// Ptr_net_58995670778129
// MI455X (gfx1250) — compile-verified, matrix operand order checked by reading, not run on hardware
//
#include <hip/hip_runtime.h>
#include <hip/hip_bf16.h>
#include <cstddef>
#include <cstdint>

// ---------------------------------------------------------------------------
// Problem dims
// ---------------------------------------------------------------------------
#define HH   230          // hidden
#define DD   460          // 2H feature dim
#define PP   512          // passage length
#define QQ   64           // question length
#define BB   32           // batch
#define DP   480          // D padded to mult of 32 (attention K)
#define KIN  1856         // 8H=1840 padded to mult of 32 (LSTM input K)
#define NG   928          // 4H=920 padded to 16 (gate dim)
#define KH   256          // H=230 padded to mult of 32 (recurrent K)
#define EP   464          // 2H padded to 16 (scorer N)
#define EK   480          // 2H padded to 32 (scorer K)
#define MROWS (BB*PP)     // 16384

typedef __attribute__((ext_vector_type(2)))  float  v2f;
typedef __attribute__((ext_vector_type(8)))  float  v8f;
typedef __attribute__((ext_vector_type(8)))  __bf16 v8bf;
typedef __attribute__((ext_vector_type(16))) __bf16 v16bf;

#define DEV static __device__ __forceinline__

DEV float sigf(float x) { return 1.0f / (1.0f + __expf(-x)); }

DEV v16bf pack16(v8bf lo, v8bf hi) {
  v16bf r;
#pragma unroll
  for (int i = 0; i < 8; ++i) { r[i] = lo[i]; r[i + 8] = hi[i]; }
  return r;
}

// One bf16 WMMA k-step: A row (16 lanes = 16 M rows), B row (16 lanes = 16 N cols),
// per-lane K chunks {base..base+7, base+16..base+23}, base = (lane<16)?0:8.
DEV v8f wmma_bf16_step(const __bf16* arow, const __bf16* brow, int k0, int base, v8f acc) {
  v8bf alo = *(const v8bf*)(arow + k0 + base);
  v8bf ahi = *(const v8bf*)(arow + k0 + 16 + base);
  v8bf blo = *(const v8bf*)(brow + k0 + base);
  v8bf bhi = *(const v8bf*)(brow + k0 + 16 + base);
  return __builtin_amdgcn_wmma_f32_16x16x32_bf16(false, pack16(alo, ahi),
                                                 false, pack16(blo, bhi),
                                                 (short)0, acc, false, false);
}

// ---------------------------------------------------------------------------
// Async global -> LDS copy path (CDNA5 ASYNCcnt); guarded by __has_builtin with
// a synchronous LDS-staging fallback so the file compiles on any toolchain.
// Builtin prototype (from hipcc diagnostics): param0 = v4i in AS1 (global),
// param1 = LDS dest, then imm offset + cpol.
// ---------------------------------------------------------------------------
#if defined(__AMDGCN__) && __has_builtin(__builtin_amdgcn_global_load_async_to_lds_b128) && \
    __has_builtin(__builtin_amdgcn_s_wait_asynccnt)
#define ASYNC_LDS 1
typedef __attribute__((vector_size(16))) int v4i;
typedef __attribute__((address_space(1))) v4i as1_v4i;
typedef __attribute__((address_space(3))) v4i as3_v4i;
DEV void async_cp16(const __bf16* g, __bf16* l) {
  __builtin_amdgcn_global_load_async_to_lds_b128((as1_v4i*)g, (as3_v4i*)l, 0, 0);
}
DEV void async_wait0() { __builtin_amdgcn_s_wait_asynccnt(0); }
#else
#define ASYNC_LDS 0
#endif

// ---------------------------------------------------------------------------
// 1) Transposed, scaled copies for the attention f32 GEMMs
//    pwT[b][d][p] = p_features[p][b][d] * w_pq[d]   (d >= 460 -> 0)
//    qT [b][d][q] = q_features[q][b][d]             (d >= 460 -> 0)
// ---------------------------------------------------------------------------
__global__ void build_pwT(const float* __restrict__ pfeat,
                          const float* __restrict__ score_w,
                          float* __restrict__ pwT) {
  __shared__ float tile[16][17];
  int b = blockIdx.y, d0 = blockIdx.x * 16;
  int tx = threadIdx.x, ty = threadIdx.y;
  for (int p0 = 0; p0 < PP; p0 += 16) {
    int d = d0 + tx;
    float v = (d < DD) ? pfeat[((size_t)(p0 + ty) * BB + b) * DD + d] : 0.f;
    tile[ty][tx] = v;
    __syncthreads();
    int dd = d0 + ty;
    float w = (dd < DD) ? score_w[2 * DD + dd] : 0.f;    // w_pq
    pwT[((size_t)b * DP + dd) * PP + p0 + tx] = tile[tx][ty] * w;
    __syncthreads();
  }
}

__global__ void build_qT(const float* __restrict__ qfeat,
                         float* __restrict__ qT) {
  __shared__ float tile[16][17];
  int b = blockIdx.y, d0 = blockIdx.x * 16;
  int tx = threadIdx.x, ty = threadIdx.y;
  for (int q0 = 0; q0 < QQ; q0 += 16) {
    int d = d0 + tx;
    float v = (d < DD) ? qfeat[((size_t)(q0 + ty) * BB + b) * DD + d] : 0.f;
    tile[ty][tx] = v;
    __syncthreads();
    int dd = d0 + ty;
    qT[((size_t)b * DP + dd) * QQ + q0 + tx] = (dd < DD) ? tile[tx][ty] : 0.f;
    __syncthreads();
  }
}

// ---------------------------------------------------------------------------
// 2) Rank-1 terms: pb[b*512+p] = p.w_p ; qb[b*64+q] = q.w_q   (wave per row)
// ---------------------------------------------------------------------------
__global__ void row_dots(const float* __restrict__ pfeat, const float* __restrict__ qfeat,
                         const float* __restrict__ score_w,
                         float* __restrict__ pb, float* __restrict__ qb) {
  int wid = (blockIdx.x * blockDim.x + threadIdx.x) >> 5;
  int lane = threadIdx.x & 31;
  if (wid < MROWS) {
    int b = wid >> 9, p = wid & 511;
    const float* row = pfeat + ((size_t)p * BB + b) * DD;
    float a = 0.f;
    for (int d = lane; d < DD; d += 32) a += row[d] * score_w[d];
    for (int o = 16; o; o >>= 1) a += __shfl_xor(a, o, 32);
    if (!lane) pb[wid] = a;
  } else if (wid < MROWS + BB * QQ) {
    int m = wid - MROWS;
    int b = m >> 6, q = m & 63;
    const float* row = qfeat + ((size_t)q * BB + b) * DD;
    float a = 0.f;
    for (int d = lane; d < DD; d += 32) a += row[d] * score_w[DD + d];
    for (int o = 16; o; o >>= 1) a += __shfl_xor(a, o, 32);
    if (!lane) qb[m] = a;
  }
}

// ---------------------------------------------------------------------------
// 3) Attention score GEMM (f32 WMMA 16x16x4): u[b][p][q] = sum_d pwT*qT
// ---------------------------------------------------------------------------
__global__ void attn_gemm(const float* __restrict__ pwT, const float* __restrict__ qT,
                          float* __restrict__ u) {
  int wid = (blockIdx.x * blockDim.x + threadIdx.x) >> 5;
  int lane = threadIdx.x & 31;
  if (wid >= 32 * 4 * BB) return;
  int qt = wid & 3, pt = (wid >> 2) & 31, b = wid >> 7;
  int half = lane >> 4, l = lane & 15;
  int p0 = pt * 16, q0 = qt * 16;
  const float* pw = pwT + (size_t)b * DP * PP;
  const float* qv = qT + (size_t)b * DP * QQ;
  v8f acc = {};
  for (int k0 = 0; k0 < DP; k0 += 4) {
    int ka = k0 + half * 2;            // A lanes 16-31 hold K=2,3
    v2f a, bb;
    a[0]  = pw[(size_t)ka * PP + p0 + l];
    a[1]  = pw[(size_t)(ka + 1) * PP + p0 + l];
    bb[0] = qv[(size_t)ka * QQ + q0 + l];
    bb[1] = qv[(size_t)(ka + 1) * QQ + q0 + l];
    acc = __builtin_amdgcn_wmma_f32_16x16x4_f32(false, a, false, bb, (short)0, acc, false, false);
  }
  float* ub = u + (size_t)b * PP * QQ;
#pragma unroll
  for (int r = 0; r < 8; ++r) {
    int m = p0 + r + half * 8;
    ub[(size_t)m * QQ + q0 + l] = acc[r];
  }
}

// ---------------------------------------------------------------------------
// 4) Add rank-1 terms + bias, softmax over Q (in place -> q2c) and c2q = max
// ---------------------------------------------------------------------------
__global__ void softmax_q(float* __restrict__ u, const float* __restrict__ pb,
                          const float* __restrict__ qb, const float* __restrict__ score_b,
                          float* __restrict__ c2q) {
  int wid = (blockIdx.x * blockDim.x + threadIdx.x) >> 5;
  int lane = threadIdx.x & 31;
  if (wid >= MROWS) return;
  int b = wid >> 9;
  float* row = u + (size_t)wid * QQ;
  float sb = score_b[0], pv = pb[wid];
  float v0 = row[lane]      + pv + qb[b * QQ + lane]      + sb;
  float v1 = row[lane + 32] + pv + qb[b * QQ + lane + 32] + sb;
  float mx = fmaxf(v0, v1);
  for (int o = 16; o; o >>= 1) mx = fmaxf(mx, __shfl_xor(mx, o, 32));
  float e0 = __expf(v0 - mx), e1 = __expf(v1 - mx);
  float sm = e0 + e1;
  for (int o = 16; o; o >>= 1) sm += __shfl_xor(sm, o, 32);
  row[lane] = e0 / sm;
  row[lane + 32] = e1 / sm;
  if (!lane) c2q[wid] = mx;
}

// ---------------------------------------------------------------------------
// 5) new_u GEMM (f32 WMMA 16x16x4): new_u[b][p][d] = sum_q q2c[b][p][q]*q_t[b][q][d]
// ---------------------------------------------------------------------------
__global__ void newu_gemm(const float* __restrict__ q2c, const float* __restrict__ qfeat,
                          float* __restrict__ new_u) {
  int wid = (blockIdx.x * blockDim.x + threadIdx.x) >> 5;
  int lane = threadIdx.x & 31;
  if (wid >= 32 * 29 * BB) return;
  int dt = wid % 29;
  int rest = wid / 29;
  int pt = rest & 31, b = rest >> 5;
  int half = lane >> 4, l = lane & 15;
  int p0 = pt * 16, d0 = dt * 16;
  int d = d0 + l;
  bool din = (d < DD);
  v8f acc = {};
  for (int k0 = 0; k0 < QQ; k0 += 4) {
    int ka = k0 + half * 2;
    v2f a, bb;
    const float* arow = q2c + ((size_t)b * PP + p0 + l) * QQ;
    a[0] = arow[ka];
    a[1] = arow[ka + 1];
    bb[0] = din ? qfeat[((size_t)ka * BB + b) * DD + d] : 0.f;
    bb[1] = din ? qfeat[((size_t)(ka + 1) * BB + b) * DD + d] : 0.f;
    acc = __builtin_amdgcn_wmma_f32_16x16x4_f32(false, a, false, bb, (short)0, acc, false, false);
  }
  if (din) {
#pragma unroll
    for (int r = 0; r < 8; ++r) {
      int m = p0 + r + half * 8;
      new_u[((size_t)b * PP + m) * DD + d] = acc[r];
    }
  }
}

// ---------------------------------------------------------------------------
// 6) Assemble final_in (bf16, 16384 x 1856) row = b*512+p
// ---------------------------------------------------------------------------
__global__ void assemble_x(const float* __restrict__ pfeat, const float* __restrict__ new_u,
                           const float* __restrict__ c2q, __bf16* __restrict__ X) {
  int row = blockIdx.x;
  int b = row >> 9, p = row & 511;
  const float* pf = pfeat + ((size_t)p * BB + b) * DD;
  const float* nu = new_u + (size_t)row * DD;
  float cq = c2q[row];
  __bf16* xr = X + (size_t)row * KIN;
  for (int d = threadIdx.x; d < KIN; d += blockDim.x) {
    float v = 0.f;
    if (d < DD)            v = pf[d];
    else if (d < 2 * DD)   v = cq * pf[d - DD];
    else if (d < 3 * DD)   v = pf[d - 2 * DD] * nu[d - 2 * DD];
    else if (d < 4 * DD) { float x = pf[d - 3 * DD]; v = x * cq * x; }
    xr[d] = (__bf16)v;
  }
}

// ---------------------------------------------------------------------------
// 7) Generic f32 -> padded bf16 matrix convert (zero-pad rows/cols)
// ---------------------------------------------------------------------------
__global__ void cvt_pad_bf16(const float* __restrict__ src, __bf16* __restrict__ dst,
                             int srows, int scols, int sstride, int drows, int dcols) {
  int idx = blockIdx.x * blockDim.x + threadIdx.x;
  if (idx >= drows * dcols) return;
  int r = idx / dcols, c = idx % dcols;
  float v = (r < srows && c < scols) ? src[(size_t)r * sstride + c] : 0.f;
  dst[idx] = (__bf16)v;
}

__global__ void sum_bias(const float* __restrict__ b1, const float* __restrict__ b2,
                         float* __restrict__ out) {
  int n = blockIdx.x * blockDim.x + threadIdx.x;
  if (n < NG) out[n] = (n < 4 * HH) ? (b1[n] + b2[n]) : 0.f;
}

// ---------------------------------------------------------------------------
// 8) Big input-projection GEMM (bf16 WMMA), LDS-tiled + double-buffered async
//    global->LDS copies. Macro-tile 64x32 per 256-thread block (8 waves, one
//    16x16 tile each; A reused 2x, B reused 4x). 256 M-blk x 29 N-blk.
//    gates_x[(p*32+b)*928+n] = X @ Wih^T, K = 1856 (58 stages of 32).
// ---------------------------------------------------------------------------
#define GG_KS (KIN / 32)    // 58 stages

__global__ __launch_bounds__(256) void gemm_gates(const __bf16* __restrict__ X,
                                                  const __bf16* __restrict__ W,
                                                  float* __restrict__ gates) {
  __shared__ __align__(16) __bf16 aT[2][64 * 32];
  __shared__ __align__(16) __bf16 bT[2][32 * 32];
  int tid = threadIdx.x, lane = tid & 31, w = tid >> 5;
  int half = lane >> 4, l = lane & 15, base = half * 8;
  int mb = blockIdx.x / 29, nb = blockIdx.x % 29;
  int m0 = mb * 64, n0 = nb * 32;
  int mt = w >> 1, nt = w & 1;

  // copy coordinates: A = 64 rows x 4 x 8-elem chunks, B = 32 rows x 4 chunks
  int ar = tid >> 2, ac = (tid & 3) * 8;
  int br = (tid & 127) >> 2, bc = ac;
  const __bf16* ag = X + (size_t)(m0 + ar) * KIN + ac;
  const __bf16* bg = W + (size_t)(n0 + br) * KIN + bc;
  __bf16* al = &aT[0][ar * 32 + ac];
  __bf16* bl = &bT[0][br * 32 + bc];
  const int lstride = 64 * 32;  // aT buffer stride; bT stride is 32*32

#if ASYNC_LDS
  async_cp16(ag, al);
  if (tid < 128) async_cp16(bg, bl);
#endif

  v8f acc = {};
  for (int s = 0; s < GG_KS; ++s) {
    int buf = s & 1;
#if ASYNC_LDS
    async_wait0();
    __syncthreads();   // stage s visible; all waves done reading stage s-1
    if (s + 1 < GG_KS) {
      int k1 = (s + 1) * 32;
      async_cp16(ag + k1, al + (buf ^ 1) * lstride);
      if (tid < 128) async_cp16(bg + k1, bl + (buf ^ 1) * (32 * 32));
    }
#else
    int k0 = s * 32;
    __syncthreads();
    *(v8bf*)(al + buf * lstride) = *(const v8bf*)(ag + k0);
    if (tid < 128) *(v8bf*)(bl + buf * (32 * 32)) = *(const v8bf*)(bg + k0);
    __syncthreads();
#endif
    const __bf16* arow = &aT[buf][(mt * 16 + l) * 32];
    const __bf16* brow = &bT[buf][(nt * 16 + l) * 32];
    acc = wmma_bf16_step(arow, brow, 0, base, acc);
  }

  int n = n0 + nt * 16 + l;
#pragma unroll
  for (int r = 0; r < 8; ++r) {
    int m = m0 + mt * 16 + r + half * 8;
    int b = m >> 9, p = m & 511;
    gates[((size_t)p * BB + b) * NG + n] = acc[r];
  }
}

// ---------------------------------------------------------------------------
// 9) LSTM recurrence: one persistent workgroup (32 waves), 512 sequential steps.
//    h (bf16) lives in LDS, c (f32) in registers, gates round-trip through LDS.
//    Recurrent matmul: 2x58 WMMA tiles per step, K = 256 (8 k-steps).
// ---------------------------------------------------------------------------
#define SMEM_G   (BB * NG * 4)        // 118784 B  f32 gates
#define SMEM_H   (BB * KH * 2)        // 16384 B   bf16 h
#define SMEM_REC (SMEM_G + SMEM_H)    // 135168 B

template <int REVERSE>
__global__ __launch_bounds__(1024) void lstm_recur(const float* __restrict__ gates_x,
                                                   const __bf16* __restrict__ Whh,
                                                   const float* __restrict__ bias,
                                                   float* __restrict__ outb) {
  extern __shared__ char smem[];
  float*  g_lds = (float*)smem;
  __bf16* h_lds = (__bf16*)(smem + SMEM_G);
  int tid = threadIdx.x, lane = tid & 31, w = tid >> 5;
  int half = lane >> 4, l = lane & 15, base = half * 8;

  for (int i = tid; i < BB * KH; i += 1024) h_lds[i] = (__bf16)0.f;

  int bs[8], js[8];
  float c_reg[8];
#pragma unroll
  for (int k = 0; k < 8; ++k) {
    int idx = tid + k * 1024;
    bs[k] = idx / HH; js[k] = idx % HH; c_reg[k] = 0.f;
  }
  __syncthreads();

  for (int t = 0; t < PP; ++t) {
    int p = REVERSE ? (PP - 1 - t) : t;
    const float* gx = gates_x + (size_t)p * BB * NG;

    for (int tile = w; tile < 2 * 58; tile += 32) {
      int mt = tile / 58, nt = tile % 58;
      const __bf16* arow = h_lds + (mt * 16 + l) * KH;
      const __bf16* brow = Whh + (size_t)(nt * 16 + l) * KH;
      v8f acc = {};
#pragma unroll
      for (int k0 = 0; k0 < KH; k0 += 32)
        acc = wmma_bf16_step(arow, brow, k0, base, acc);
      int n = nt * 16 + l;
      float bv = bias[n];
#pragma unroll
      for (int r = 0; r < 8; ++r) {
        int b = mt * 16 + r + half * 8;
        g_lds[b * NG + n] = acc[r] + gx[b * NG + n] + bv;
      }
    }
    __syncthreads();

#pragma unroll
    for (int k = 0; k < 8; ++k) {
      int idx = tid + k * 1024;
      if (idx < BB * HH) {
        int b = bs[k], j = js[k];
        float gi = g_lds[b * NG + j];
        float gf = g_lds[b * NG + HH + j];
        float gg = g_lds[b * NG + 2 * HH + j];
        float go = g_lds[b * NG + 3 * HH + j];
        float cc = sigf(gf) * c_reg[k] + sigf(gi) * tanhf(gg);
        c_reg[k] = cc;
        float h = sigf(go) * tanhf(cc);
        h_lds[b * KH + j] = (__bf16)h;
        outb[((size_t)b * PP + p) * DD + j] = h;
      }
    }
    __syncthreads();
  }
}

// ---------------------------------------------------------------------------
// 10) Pointer scorer GEMM (bf16 WMMA): G[m][e] = final @ w2a_w[:, :460]^T
// ---------------------------------------------------------------------------
__global__ void gemm_ptr(const __bf16* __restrict__ X, const __bf16* __restrict__ W,
                         float* __restrict__ G) {
  int wid = (blockIdx.x * blockDim.x + threadIdx.x) >> 5;
  int lane = threadIdx.x & 31;
  if (wid >= 1024 * 29) return;
  int mt = wid / 29, nt = wid % 29;
  int half = lane >> 4, l = lane & 15, base = half * 8;
  const __bf16* arow = X + (size_t)(mt * 16 + l) * EK;
  const __bf16* brow = W + (size_t)(nt * 16 + l) * EK;
  v8f acc = {};
  for (int k0 = 0; k0 < EK; k0 += 32)
    acc = wmma_bf16_step(arow, brow, k0, base, acc);
  int n = nt * 16 + l;
#pragma unroll
  for (int r = 0; r < 8; ++r) {
    int m = mt * 16 + r + half * 8;
    G[(size_t)m * EP + n] = acc[r];
  }
}

// ---------------------------------------------------------------------------
// 11) s[m] = sum_e tanh(G[m][e] + bias[e]) * w1a[e] + w1a_b  (wave per row)
// ---------------------------------------------------------------------------
__global__ void ptr_score(const float* __restrict__ G, const float* __restrict__ bias2,
                          int per_b, const float* __restrict__ w1a,
                          const float* __restrict__ w1ab, float* __restrict__ s) {
  int m = (blockIdx.x * blockDim.x + threadIdx.x) >> 5;
  int lane = threadIdx.x & 31;
  if (m >= MROWS) return;
  const float* g = G + (size_t)m * EP;
  const float* bb = per_b ? (bias2 + (size_t)(m >> 9) * DD) : bias2;
  float acc = 0.f;
  for (int e = lane; e < DD; e += 32) acc += tanhf(g[e] + bb[e]) * w1a[e];
  for (int o = 16; o; o >>= 1) acc += __shfl_xor(acc, o, 32);
  if (!lane) s[m] = acc + w1ab[0];
}

// ---------------------------------------------------------------------------
// 12) softmax over P (block per batch row); writes log-softmax, optionally alpha
// ---------------------------------------------------------------------------
__global__ void softmax_p(const float* __restrict__ s, float* __restrict__ log_out,
                          float* __restrict__ alpha_out) {
  __shared__ float red[512];
  int b = blockIdx.x, t = threadIdx.x;
  float x = s[b * PP + t];
  red[t] = x; __syncthreads();
  for (int o = 256; o; o >>= 1) { if (t < o) red[t] = fmaxf(red[t], red[t + o]); __syncthreads(); }
  float mx = red[0]; __syncthreads();
  float e = __expf(x - mx);
  red[t] = e; __syncthreads();
  for (int o = 256; o; o >>= 1) { if (t < o) red[t] += red[t + o]; __syncthreads(); }
  float sm = red[0];
  log_out[b * PP + t] = x - mx - logf(sm);
  if (alpha_out) alpha_out[b * PP + t] = e / sm;
}

// 13) c1[b][d] = sum_p alpha[b][p] * final[b][p][d]
__global__ void compute_c1(const float* __restrict__ alpha, const float* __restrict__ fin,
                           float* __restrict__ c1) {
  int idx = blockIdx.x * blockDim.x + threadIdx.x;
  if (idx >= BB * DD) return;
  int b = idx / DD, d = idx % DD;
  float acc = 0.f;
  for (int p = 0; p < PP; ++p)
    acc += alpha[b * PP + p] * fin[((size_t)b * PP + p) * DD + d];
  c1[idx] = acc;
}

// 14) boundary BiLSTM (seq len 1, h0=c0=0 => no Whh term)
__global__ void h0a_cell(const float* __restrict__ c1,
                         const float* __restrict__ Wf, const float* __restrict__ bf1, const float* __restrict__ bf2,
                         const float* __restrict__ Wb, const float* __restrict__ bb1, const float* __restrict__ bb2,
                         float* __restrict__ h0a) {
  int idx = blockIdx.x * blockDim.x + threadIdx.x;
  if (idx >= 2 * BB * HH) return;
  int dir = idx / (BB * HH);
  int rr = idx % (BB * HH);
  int b = rr / HH, j = rr % HH;
  const float* W  = dir ? Wb : Wf;
  const float* b1 = dir ? bb1 : bf1;
  const float* b2 = dir ? bb2 : bf2;
  const float* x = c1 + (size_t)b * DD;
  float g4[4];
#pragma unroll
  for (int gi = 0; gi < 4; ++gi) {
    int n = gi * HH + j;
    float a = b1[n] + b2[n];
    const float* wr = W + (size_t)n * DD;
    for (int d = 0; d < DD; ++d) a += x[d] * wr[d];
    g4[gi] = a;
  }
  float cc = sigf(g4[0]) * tanhf(g4[2]);     // f-gate * c0 = 0
  float h = sigf(g4[3]) * tanhf(cc);
  h0a[(size_t)b * DD + dir * HH + j] = h;
}

// 15) bias2[b][e] = w2a_b[e] + sum_d h0a[b][d] * w2a_w[e][460+d]
__global__ void build_bias2(const float* __restrict__ h0a, const float* __restrict__ w2a_w,
                            const float* __restrict__ w2a_b, float* __restrict__ bias2) {
  int idx = blockIdx.x * blockDim.x + threadIdx.x;
  if (idx >= BB * DD) return;
  int b = idx / DD, e = idx % DD;
  float a = w2a_b[e];
  const float* wr = w2a_w + (size_t)e * (2 * DD) + DD;
  const float* hr = h0a + (size_t)b * DD;
  for (int d = 0; d < DD; ++d) a += hr[d] * wr[d];
  bias2[idx] = a;
}

// ---------------------------------------------------------------------------
// Workspace layout (bytes, 256-aligned). Aliasing: PWT region -> new_u -> G;
// FIN region -> final_bf; WIH region -> W1bf; one gates buffer reused per dir.
// Total ~158 MB.
// ---------------------------------------------------------------------------
#define OFS_PWT   ((size_t)0)                       // 31457280  (pwT | new_u | G)
#define OFS_QT    ((size_t)31457280)                // 3932160
#define OFS_U     ((size_t)35389440)                // 4194304   (u | q2c)
#define OFS_PB    ((size_t)39583744)                // 65536
#define OFS_QB    ((size_t)39649280)                // 8192
#define OFS_C2Q   ((size_t)39657472)                // 65536
#define OFS_FIN   ((size_t)39723008)                // 60817408  (final_in | final_bf)
#define OFS_WIH   ((size_t)100540416)               // 3444736   (Wih_bf | W1bf)
#define OFS_GATES ((size_t)103985152)               // 60817408
#define OFS_WHH   ((size_t)164802560)               // 475136
#define OFS_BIAS  ((size_t)165277696)               // 4096
#define OFS_S     ((size_t)165281792)               // 65536
#define OFS_ALPHA ((size_t)165347328)               // 65536
#define OFS_C1    ((size_t)165412864)               // 61440
#define OFS_H0A   ((size_t)165474304)               // 61440
#define OFS_B2    ((size_t)165535744)               // 61440

extern "C" void kernel_launch(void* const* d_in, const int* in_sizes, int n_in,
                              void* d_out, int out_size, void* d_ws, size_t ws_size,
                              hipStream_t stream) {
  (void)in_sizes; (void)n_in; (void)out_size; (void)ws_size;
  const float* pfeat   = (const float*)d_in[0];
  const float* qfeat   = (const float*)d_in[1];
  const float* score_w = (const float*)d_in[2];
  const float* score_b = (const float*)d_in[3];
  const float* m_Wih_f = (const float*)d_in[4];
  const float* m_Whh_f = (const float*)d_in[5];
  const float* m_bih_f = (const float*)d_in[6];
  const float* m_bhh_f = (const float*)d_in[7];
  const float* m_Wih_b = (const float*)d_in[8];
  const float* m_Whh_b = (const float*)d_in[9];
  const float* m_bih_b = (const float*)d_in[10];
  const float* m_bhh_b = (const float*)d_in[11];
  const float* w2a_w   = (const float*)d_in[12];
  const float* w2a_b   = (const float*)d_in[13];
  const float* w1a_w   = (const float*)d_in[14];
  const float* w1a_b   = (const float*)d_in[15];
  const float* b_Wih_f = (const float*)d_in[16];
  const float* b_bih_f = (const float*)d_in[18];
  const float* b_bhh_f = (const float*)d_in[19];
  const float* b_Wih_b = (const float*)d_in[20];
  const float* b_bih_b = (const float*)d_in[22];
  const float* b_bhh_b = (const float*)d_in[23];

  char* ws = (char*)d_ws;
  float*  pwT     = (float*)(ws + OFS_PWT);
  float*  new_u   = (float*)(ws + OFS_PWT);   // alias
  float*  G       = (float*)(ws + OFS_PWT);   // alias
  float*  qT      = (float*)(ws + OFS_QT);
  float*  u       = (float*)(ws + OFS_U);
  float*  pb      = (float*)(ws + OFS_PB);
  float*  qb      = (float*)(ws + OFS_QB);
  float*  c2q     = (float*)(ws + OFS_C2Q);
  __bf16* Xin     = (__bf16*)(ws + OFS_FIN);
  __bf16* fin_bf  = (__bf16*)(ws + OFS_FIN);  // alias
  __bf16* Wih_bf  = (__bf16*)(ws + OFS_WIH);
  __bf16* W1_bf   = (__bf16*)(ws + OFS_WIH);  // alias
  float*  gates   = (float*)(ws + OFS_GATES);
  __bf16* Whh_bf  = (__bf16*)(ws + OFS_WHH);
  float*  bias    = (float*)(ws + OFS_BIAS);
  float*  sbuf    = (float*)(ws + OFS_S);
  float*  alpha   = (float*)(ws + OFS_ALPHA);
  float*  c1      = (float*)(ws + OFS_C1);
  float*  h0a     = (float*)(ws + OFS_H0A);
  float*  bias2   = (float*)(ws + OFS_B2);

  float* out_final = (float*)d_out;                 // (B,P,460)
  float* out_la1   = out_final + (size_t)BB * PP * DD;
  float* out_la2   = out_la1 + BB * PP;

  (void)hipFuncSetAttribute(reinterpret_cast<const void*>(lstm_recur<0>),
                            hipFuncAttributeMaxDynamicSharedMemorySize, SMEM_REC);
  (void)hipFuncSetAttribute(reinterpret_cast<const void*>(lstm_recur<1>),
                            hipFuncAttributeMaxDynamicSharedMemorySize, SMEM_REC);

  // --- attention ---
  build_pwT<<<dim3(DP / 16, BB), dim3(16, 16), 0, stream>>>(pfeat, score_w, pwT);
  build_qT<<<dim3(DP / 16, BB), dim3(16, 16), 0, stream>>>(qfeat, qT);
  row_dots<<<(MROWS + BB * QQ) * 32 / 256, 256, 0, stream>>>(pfeat, qfeat, score_w, pb, qb);
  attn_gemm<<<32 * 4 * BB / 8, 256, 0, stream>>>(pwT, qT, u);
  softmax_q<<<MROWS * 32 / 256, 256, 0, stream>>>(u, pb, qb, score_b, c2q);
  newu_gemm<<<32 * 29 * BB / 8, 256, 0, stream>>>(u, qfeat, new_u);
  assemble_x<<<MROWS, 256, 0, stream>>>(pfeat, new_u, c2q, Xin);

  // --- BiLSTM forward direction ---
  cvt_pad_bf16<<<(NG * KIN + 255) / 256, 256, 0, stream>>>(m_Wih_f, Wih_bf, 4 * HH, 8 * HH, 8 * HH, NG, KIN);
  cvt_pad_bf16<<<(NG * KH + 255) / 256, 256, 0, stream>>>(m_Whh_f, Whh_bf, 4 * HH, HH, HH, NG, KH);
  sum_bias<<<(NG + 255) / 256, 256, 0, stream>>>(m_bih_f, m_bhh_f, bias);
  gemm_gates<<<256 * 29, 256, 0, stream>>>(Xin, Wih_bf, gates);
  lstm_recur<0><<<1, 1024, SMEM_REC, stream>>>(gates, Whh_bf, bias, out_final);

  // --- BiLSTM backward direction (reuses Wih/gates buffers; stream-ordered) ---
  cvt_pad_bf16<<<(NG * KIN + 255) / 256, 256, 0, stream>>>(m_Wih_b, Wih_bf, 4 * HH, 8 * HH, 8 * HH, NG, KIN);
  cvt_pad_bf16<<<(NG * KH + 255) / 256, 256, 0, stream>>>(m_Whh_b, Whh_bf, 4 * HH, HH, HH, NG, KH);
  sum_bias<<<(NG + 255) / 256, 256, 0, stream>>>(m_bih_b, m_bhh_b, bias);
  gemm_gates<<<256 * 29, 256, 0, stream>>>(Xin, Wih_bf, gates);
  lstm_recur<1><<<1, 1024, SMEM_REC, stream>>>(gates, Whh_bf, bias, out_final + HH);

  // --- pointer scorer (G shared by both calls) ---
  cvt_pad_bf16<<<(MROWS * EK + 255) / 256, 256, 0, stream>>>(out_final, fin_bf, MROWS, DD, DD, MROWS, EK);
  cvt_pad_bf16<<<(EP * EK + 255) / 256, 256, 0, stream>>>(w2a_w, W1_bf, DD, DD, 2 * DD, EP, EK);
  gemm_ptr<<<1024 * 29 / 8, 256, 0, stream>>>(fin_bf, W1_bf, G);

  // call 1: h0a = 0 -> bias is just w2a_b
  ptr_score<<<MROWS * 32 / 256, 256, 0, stream>>>(G, w2a_b, 0, w1a_w, w1a_b, sbuf);
  softmax_p<<<BB, PP, 0, stream>>>(sbuf, out_la1, alpha);
  compute_c1<<<(BB * DD + 255) / 256, 256, 0, stream>>>(alpha, out_final, c1);
  h0a_cell<<<(2 * BB * HH + 255) / 256, 256, 0, stream>>>(c1, b_Wih_f, b_bih_f, b_bhh_f,
                                                          b_Wih_b, b_bih_b, b_bhh_b, h0a);
  build_bias2<<<(BB * DD + 255) / 256, 256, 0, stream>>>(h0a, w2a_w, w2a_b, bias2);

  // call 2
  ptr_score<<<MROWS * 32 / 256, 256, 0, stream>>>(G, bias2, 1, w1a_w, w1a_b, sbuf);
  softmax_p<<<BB, PP, 0, stream>>>(sbuf, out_la2, nullptr);
}
